// MLABlock_42520176231023
// MI455X (gfx1250) — matrix operand order repaired and checked by reading, not compiled, not run on hardware
//
#include <hip/hip_runtime.h>
#include <math.h>

// ---------------------------------------------------------------------------
// CDNA5 (gfx1250, wave32) attention block:
//   fp8-roundtrip quant (exact e4m3 grid) -> bf16 dequant
//   bf16 WMMA GEMM (QKV) -> bf16 WMMA flash-attention -> bf16 WMMA GEMM
// Tile staging uses GLOBAL_LOAD_ASYNC_TO_LDS_B128 (ASYNCcnt) + double buffering.
// ---------------------------------------------------------------------------

typedef __bf16 bf16;
typedef __attribute__((ext_vector_type(16))) __bf16 v16bf;
typedef __attribute__((ext_vector_type(8)))  __bf16 bf16x8;
typedef __attribute__((ext_vector_type(8)))  float  v8f;
typedef __attribute__((ext_vector_type(4)))  float  f32x4;

__device__ __forceinline__ v16bf combine16(bf16x8 lo, bf16x8 hi) {
    return __builtin_shufflevector(lo, hi, 0,1,2,3,4,5,6,7,8,9,10,11,12,13,14,15);
}

__device__ __forceinline__ v8f wmma_bf16(v16bf a, v16bf b, v8f c) {
    // D = A(16x32) * B(32x16) + C, f32 accumulate
    return __builtin_amdgcn_wmma_f32_16x16x32_bf16(
        false, a, false, b, (short)0, c, false, false);
}

// Async DMA: copy 16 bytes per lane from global into LDS (per-lane LDS address).
// Low 32 bits of a flat shared-aperture address are the LDS byte offset.
__device__ __forceinline__ void async_copy_b128(void* lds, const void* gmem) {
    unsigned int l = (unsigned int)(unsigned long long)(uintptr_t)lds;
    asm volatile("global_load_async_to_lds_b128 %0, %1, off"
                 :: "v"(l), "v"(gmem) : "memory");
}
__device__ __forceinline__ void wait_async() {
    asm volatile("s_wait_asynccnt 0x0" ::: "memory");
}

// Exact e4m3(fn) lattice, branchless: step = max(2^(floor(log2 a)-3), 2^-9)
// covers normals (1+3-bit mantissa), subnormals (grid 2^-9) and saturation.
__device__ __forceinline__ float e4m3_roundtrip(float x) {
    float a = fminf(fabsf(x), 448.0f);
    unsigned int eb = __float_as_uint(a) & 0x7f800000u;   // 2^floor(log2 a)
    float step = fmaxf(__uint_as_float(eb) * 0.125f, 0x1p-9f);
    float q = rintf(a / step) * step;                     // RNE onto the grid
    return copysignf(q, x);
}

// ---------------------------------------------------------------------------
// Block-quantize (128 along C) + dequantize to bf16. One wave per block.
// transpose=0: out[R][C]; transpose=1: out[C][R] (GEMM B operand layout).
// ---------------------------------------------------------------------------
__global__ __launch_bounds__(256)
void quant_rt_kernel(const float* __restrict__ in, bf16* __restrict__ out,
                     int R, int C, int transpose) {
    int gwave = (int)((blockIdx.x * blockDim.x + threadIdx.x) >> 5);
    int lane  = threadIdx.x & 31;
    int nbC   = C >> 7;
    if (gwave >= R * nbC) return;
    int row = gwave / nbC, cb = gwave % nbC;

    f32x4 v = *(const f32x4*)(in + (size_t)row * C + cb * 128 + lane * 4);
    float m = fmaxf(fmaxf(fabsf(v.x), fabsf(v.y)), fmaxf(fabsf(v.z), fabsf(v.w)));
#pragma unroll
    for (int off = 16; off; off >>= 1) m = fmaxf(m, __shfl_xor(m, off, 32));
    float s   = fmaxf(m * (1.0f / 448.0f), 1e-12f);
    float inv = 1.0f / s;

    float r0 = e4m3_roundtrip(v.x * inv) * s;
    float r1 = e4m3_roundtrip(v.y * inv) * s;
    float r2 = e4m3_roundtrip(v.z * inv) * s;
    float r3 = e4m3_roundtrip(v.w * inv) * s;

    if (!transpose) {
        bf16 o4[4] = {(bf16)r0, (bf16)r1, (bf16)r2, (bf16)r3};
        *(unsigned long long*)(out + (size_t)row * C + cb * 128 + lane * 4) =
            *(const unsigned long long*)o4;
    } else {
        int c0 = cb * 128 + lane * 4;
        out[(size_t)(c0 + 0) * R + row] = (bf16)r0;
        out[(size_t)(c0 + 1) * R + row] = (bf16)r1;
        out[(size_t)(c0 + 2) * R + row] = (bf16)r2;
        out[(size_t)(c0 + 3) * R + row] = (bf16)r3;
    }
}

// ---------------------------------------------------------------------------
// bf16 GEMM: C[M][N] = A[M][K] @ Bt[N][K]^T + bias[N]
// 128x128 tile, BK=32, 8 waves (4x2), wave tile 32x64 (2x4 WMMA tiles).
// Double-buffered LDS fed by async-to-LDS DMA.
// ---------------------------------------------------------------------------
template <bool OUT_BF16>
__global__ __launch_bounds__(256)
void gemm_bf16_kernel(const bf16* __restrict__ A, const bf16* __restrict__ Bt,
                      const float* __restrict__ bias, void* __restrict__ Cout,
                      int M, int N, int K) {
    constexpr int BM = 128, BN = 128, BK = 32, LDT = 40;  // rows padded to 80B
    __shared__ bf16 As[2][BM][LDT];
    __shared__ bf16 Bs[2][BN][LDT];

    const int tid  = threadIdx.x;
    const int wave = tid >> 5, lane = tid & 31;
    const int half = (lane >> 4) & 1, l16 = lane & 15;
    const int wm = wave & 3, wn = wave >> 2;
    const int m0 = blockIdx.y * BM, n0 = blockIdx.x * BN;

    v8f acc[2][4];
#pragma unroll
    for (int i = 0; i < 2; ++i)
#pragma unroll
        for (int j = 0; j < 4; ++j)
            acc[i][j] = (v8f){0.f, 0.f, 0.f, 0.f, 0.f, 0.f, 0.f, 0.f};

    const int lr = tid >> 1, lc = (tid & 1) * 16;   // per-thread 32B of A and B

    auto stage = [&](int b, int k0) {
        const bf16* ap = A  + (size_t)(m0 + lr) * K + k0 + lc;
        const bf16* bp = Bt + (size_t)(n0 + lr) * K + k0 + lc;
        async_copy_b128(&As[b][lr][lc],     ap);
        async_copy_b128(&As[b][lr][lc + 8], ap + 8);
        async_copy_b128(&Bs[b][lr][lc],     bp);
        async_copy_b128(&Bs[b][lr][lc + 8], bp + 8);
    };

    stage(0, 0);
    int buf = 0;
    for (int k0 = 0; k0 < K; k0 += BK, buf ^= 1) {
        wait_async();        // this wave's DMA slice of tile t has landed
        __syncthreads();     // => every wave's slice visible; prev reads done
        if (k0 + BK < K) stage(buf ^ 1, k0 + BK);   // overlap next tile DMA

        v16bf afrag[2], bfrag[4];
#pragma unroll
        for (int i = 0; i < 2; ++i) {   // A frag: lane l16 = row, K split by half
            const bf16* p = &As[buf][wm * 32 + i * 16 + l16][0];
            bf16x8 lo = *(const bf16x8*)(p + half * 8);
            bf16x8 hi = *(const bf16x8*)(p + 16 + half * 8);
            afrag[i] = combine16(lo, hi);
        }
#pragma unroll
        for (int j = 0; j < 4; ++j) {   // B frag: lane l16 = col, K split by half
            const bf16* p = &Bs[buf][wn * 64 + j * 16 + l16][0];
            bf16x8 lo = *(const bf16x8*)(p + half * 8);
            bf16x8 hi = *(const bf16x8*)(p + 16 + half * 8);
            bfrag[j] = combine16(lo, hi);
        }
#pragma unroll
        for (int i = 0; i < 2; ++i)
#pragma unroll
            for (int j = 0; j < 4; ++j)
                acc[i][j] = wmma_bf16(afrag[i], bfrag[j], acc[i][j]);
    }

#pragma unroll
    for (int j = 0; j < 4; ++j) {
        const int col = n0 + wn * 64 + j * 16 + l16;
        const float bj = bias[col];
#pragma unroll
        for (int i = 0; i < 2; ++i) {
#pragma unroll
            for (int r = 0; r < 8; ++r) {
                const int row = m0 + wm * 32 + i * 16 + half * 8 + r;
                const float v = acc[i][j][r] + bj;
                if (OUT_BF16)
                    ((bf16*)Cout)[(size_t)row * N + col] = (bf16)v;
                else
                    ((float*)Cout)[(size_t)row * N + col] = v;
            }
        }
    }
}

// ---------------------------------------------------------------------------
// Flash attention: qkv bf16 [B*S][3*1024] (q|k|v at h*64+d) -> O f32 [B*S][1024]
// Grid (S/128, H, B); 8 waves; wave owns 16 q-rows; 32-key tiles.
// Online softmax with a per-tile running max (shift-invariant; scores ~N(0,1),
// so exp stays in range with margin); per-row sums via WMMA against ones.
// Double-buffered K (async DMA) and V (manual transpose) tiles.
// ---------------------------------------------------------------------------
__global__ __launch_bounds__(256)
void attn_kernel(const bf16* __restrict__ qkv, float* __restrict__ O, int S) {
    constexpr int D = 64, ROW = 3072, KT = 32;
    constexpr int LDK = 72, LDV = 40, LDP = 32;
    __shared__ bf16 Ks[2][KT][LDK];               // K tile, [s][d] (natural)
    __shared__ bf16 Vs[2][D][LDV];                // V tile transposed, [d][s]
    __shared__ bf16 Ps[8][16][LDP];               // per-wave P staging

    const int tid  = threadIdx.x;
    const int wave = tid >> 5, lane = tid & 31;
    const int half = (lane >> 4) & 1, l16 = lane & 15;
    const int h = blockIdx.y, b = blockIdx.z;
    const int qrow0 = b * S + blockIdx.x * 128 + wave * 16;

    // Q fragments; fold the exact power-of-two score scale (1/8) into Q.
    v16bf aq[2];
#pragma unroll
    for (int c = 0; c < 2; ++c) {
        const bf16* qp = qkv + (size_t)(qrow0 + l16) * ROW + h * 64 + c * 32;
        bf16x8 lo = *(const bf16x8*)(qp + half * 8);
        bf16x8 hi = *(const bf16x8*)(qp + 16 + half * 8);
        v16bf a = combine16(lo, hi);
#pragma unroll
        for (int i = 0; i < 16; ++i) a[i] = (bf16)((float)a[i] * 0.125f);
        aq[c] = a;
    }

    v16bf ones;
#pragma unroll
    for (int i = 0; i < 16; ++i) ones[i] = (bf16)1.0f;

    v8f acc[4];
#pragma unroll
    for (int j = 0; j < 4; ++j) acc[j] = (v8f){0.f,0.f,0.f,0.f,0.f,0.f,0.f,0.f};
    float lrow[8];
#pragma unroll
    for (int r = 0; r < 8; ++r) lrow[r] = 0.f;
    float mtile = -1e30f;

    const int ss = tid >> 3, dc = (tid & 7) * 8;

    // prologue: stage tile 0
    {
        const size_t rbase = (size_t)(b * S + ss) * ROW + h * 64 + dc;
        async_copy_b128(&Ks[0][ss][dc], qkv + 1024 + rbase);
        bf16x8 vv = *(const bf16x8*)(qkv + 2048 + rbase);
#pragma unroll
        for (int i = 0; i < 8; ++i) Vs[0][dc + i][ss] = vv[i];
    }

    int buf = 0;
    for (int s0 = 0; s0 < S; s0 += KT, buf ^= 1) {
        wait_async();        // K(t) landed (only K(t) outstanding)
        __syncthreads();     // all slices + V stores visible; t-1 reads done

        // prefetch tile t+1 into the other buffer (overlaps compute below)
        bf16x8 vnext;
        const bool more = (s0 + KT) < S;
        if (more) {
            const size_t rb2 = (size_t)(b * S + s0 + KT + ss) * ROW + h * 64 + dc;
            async_copy_b128(&Ks[buf ^ 1][ss][dc], qkv + 1024 + rb2);
            vnext = *(const bf16x8*)(qkv + 2048 + rb2);
        }

        // S-tile = (Q/8) @ K^T : two 16x16 col-tiles x 2 d-chunks
        v8f sc[2];
#pragma unroll
        for (int c = 0; c < 2; ++c) {
            v8f s = (v8f){0.f,0.f,0.f,0.f,0.f,0.f,0.f,0.f};
#pragma unroll
            for (int d2 = 0; d2 < 2; ++d2) {
                const bf16* kp = &Ks[buf][c * 16 + l16][d2 * 32];
                bf16x8 lo = *(const bf16x8*)(kp + half * 8);
                bf16x8 hi = *(const bf16x8*)(kp + 16 + half * 8);
                s = wmma_bf16(aq[d2], combine16(lo, hi), s);
            }
            sc[c] = s;
        }

        // park V(t+1) in LDS while scores are in flight
        if (more) {
#pragma unroll
            for (int i = 0; i < 8; ++i) Vs[buf ^ 1][dc + i][ss] = vnext[i];
        }

        // online softmax, single running max per 16x32 tile
        float tm = -1e30f;
#pragma unroll
        for (int r = 0; r < 8; ++r)
            tm = fmaxf(tm, fmaxf(sc[0][r], sc[1][r]));
#pragma unroll
        for (int off = 1; off < 32; off <<= 1)
            tm = fmaxf(tm, __shfl_xor(tm, off, 32));
        const float mn   = fmaxf(mtile, tm);
        const float corr = __expf(mtile - mn);
        mtile = mn;
#pragma unroll
        for (int r = 0; r < 8; ++r) lrow[r] *= corr;
#pragma unroll
        for (int j = 0; j < 4; ++j) acc[j] *= corr;       // uniform rescale

#pragma unroll
        for (int r = 0; r < 8; ++r) {
            float p0 = __expf(sc[0][r] - mn), p1 = __expf(sc[1][r] - mn);
            Ps[wave][half * 8 + r][l16]      = (bf16)p0;
            Ps[wave][half * 8 + r][16 + l16] = (bf16)p1;
        }

        // reload P in A-layout; row sums via WMMA against ones; O += P @ V
        const bf16* pp = &Ps[wave][l16][0];
        bf16x8 plo = *(const bf16x8*)(pp + half * 8);
        bf16x8 phi = *(const bf16x8*)(pp + 16 + half * 8);
        v16bf ap = combine16(plo, phi);

        v8f rsum = wmma_bf16(ap, ones,
                             (v8f){0.f,0.f,0.f,0.f,0.f,0.f,0.f,0.f});
#pragma unroll
        for (int r = 0; r < 8; ++r) lrow[r] += rsum[r];

#pragma unroll
        for (int j = 0; j < 4; ++j) {
            const bf16* vp = &Vs[buf][j * 16 + l16][0];
            bf16x8 lo = *(const bf16x8*)(vp + half * 8);
            bf16x8 hi = *(const bf16x8*)(vp + 16 + half * 8);
            acc[j] = wmma_bf16(ap, combine16(lo, hi), acc[j]);
        }
    }

#pragma unroll
    for (int r = 0; r < 8; ++r) {
        const float inv = 1.0f / lrow[r];
        const int m = qrow0 + half * 8 + r;
#pragma unroll
        for (int j = 0; j < 4; ++j)
            O[(size_t)m * 1024 + h * 64 + j * 16 + l16] = acc[j][r] * inv;
    }
}

// ---------------------------------------------------------------------------
extern "C" void kernel_launch(void* const* d_in, const int* in_sizes, int n_in,
                              void* d_out, int out_size, void* d_ws, size_t ws_size,
                              hipStream_t stream) {
    const float* x    = (const float*)d_in[0];  // [4096][1024]
    const float* wqkv = (const float*)d_in[1];  // [1024][3072]
    const float* bqkv = (const float*)d_in[2];  // [3072]
    const float* wout = (const float*)d_in[3];  // [1024][1024]
    const float* bout = (const float*)d_in[4];  // [1024]
    float* out = (float*)d_out;                 // [4096][1024] f32

    constexpr int B = 2, S = 2048, M = B * S;   // 4096
    constexpr int D = 1024, N3 = 3072;

    char* ws = (char*)d_ws;
    bf16* xd    = (bf16*)ws;                 ws += (size_t)M * D * 2;
    bf16* wqkvT = (bf16*)ws;                 ws += (size_t)N3 * D * 2;
    bf16* woutT = (bf16*)ws;                 ws += (size_t)D * D * 2;
    bf16* qkvb  = (bf16*)ws;                 ws += (size_t)M * N3 * 2;
    float* ob   = (float*)ws;                ws += (size_t)M * D * 4;
    bf16* od    = (bf16*)ws;

    // 1) fp8 round-trip quantization -> bf16 (weights transposed for GEMM B)
    quant_rt_kernel<<<(M * (D / 128)) / 8, 256, 0, stream>>>(x, xd, M, D, 0);
    quant_rt_kernel<<<(D * (N3 / 128)) / 8, 256, 0, stream>>>(wqkv, wqkvT, D, N3, 1);
    quant_rt_kernel<<<(D * (D / 128)) / 8, 256, 0, stream>>>(wout, woutT, D, D, 1);

    // 2) QKV projection -> bf16 qkv
    gemm_bf16_kernel<true><<<dim3(N3 / 128, M / 128), 256, 0, stream>>>(
        xd, wqkvT, bqkv, qkvb, M, N3, D);

    // 3) flash attention -> f32 o
    attn_kernel<<<dim3(S / 128, 16, B), 256, 0, stream>>>(qkvb, ob, S);

    // 4) fp8 round-trip on o, then output projection -> f32 out
    quant_rt_kernel<<<(M * (D / 128)) / 8, 256, 0, stream>>>(ob, od, M, D, 0);
    gemm_bf16_kernel<false><<<dim3(D / 128, M / 128), 256, 0, stream>>>(
        od, woutT, bout, out, M, D, D);
}
